// LayerStacks_52974126629260
// MI455X (gfx1250) — matrix operand order repaired and checked by reading, not compiled, not run on hardware
//
#include <hip/hip_runtime.h>

// ---------------------------------------------------------------------------
// LayerStacks (NNUE-style) fused kernel for gfx1250.
// Stage 1: dense GEMM  X(16384x1536) @ W_all^T(1536x144)  via split-bf16 WMMA
//          (W_all = concat(W1 rows 0..127, Wf rows 0..15)), f32 accumulate.
//          A-operand split uses truncation + v_perm_b32 packing to keep the
//          VALU conversion cost under the HBM roofline.
// Stage 2: in-block tail (bucket select, activations, 64x30 matvec, 64-dot).
// ---------------------------------------------------------------------------

#define K_DIM          1536
#define N_COLS         144
#define N_TILES        9          // 144 / 16
#define M_TOTAL        16384
#define ROWS_PER_BLOCK 64         // 4 waves x 16 rows
#define LDS_STRIDE     145        // 144 + 1 pad -> conflict-free tail reads

typedef __attribute__((ext_vector_type(16))) __bf16   v16bf;
typedef __attribute__((ext_vector_type(8)))  float    v8f;
typedef __attribute__((ext_vector_type(4)))  float    v4f;
typedef __attribute__((ext_vector_type(8)))  unsigned v8u;

// pack bf16(f1)<<16 | bf16(f0), truncation rounding: one v_perm_b32
// sel 0x07060302: dest bytes = { f1.b3, f1.b2, f0.b3, f0.b2 }
#define BF16_PACK_SEL 0x07060302u

static __device__ __forceinline__ __bf16 f32_to_bf16_rne(float f) {
  unsigned u = __builtin_bit_cast(unsigned, f);
  u += 0x7FFFu + ((u >> 16) & 1u);                 // round-to-nearest-even
  unsigned short h = (unsigned short)(u >> 16);
  return __builtin_bit_cast(__bf16, h);
}
static __device__ __forceinline__ float bf16_to_f32(__bf16 b) {
  unsigned short h = __builtin_bit_cast(unsigned short, b);
  unsigned u = (unsigned)h << 16;
  return __builtin_bit_cast(float, u);
}
static __device__ __forceinline__ float clamp01(float v) {
  return fminf(fmaxf(v, 0.0f), 1.0f);
}

// --------------------------------------------------------------------------
// Pack W1 (128x1536) and Wf (16x1536) into hi/lo bf16 planes (row-major,
// contiguous K) so GEMM lanes can load 16 contiguous bf16 = one v16bf.
// One-time cost: keep RNE here for the best hi plane.
// --------------------------------------------------------------------------
__global__ void pack_weights_kernel(const float* __restrict__ W1,
                                    const float* __restrict__ Wf,
                                    unsigned short* __restrict__ whi_u,
                                    unsigned short* __restrict__ wlo_u) {
  int i = blockIdx.x * blockDim.x + threadIdx.x;
  if (i >= N_COLS * K_DIM) return;
  int n = i / K_DIM;
  int k = i - n * K_DIM;
  float w = (n < 128) ? W1[n * K_DIM + k] : Wf[(n - 128) * K_DIM + k];
  __bf16 hi = f32_to_bf16_rne(w);
  __bf16 lo = f32_to_bf16_rne(w - bf16_to_f32(hi));
  whi_u[i] = __builtin_bit_cast(unsigned short, hi);
  wlo_u[i] = __builtin_bit_cast(unsigned short, lo);
}

// --------------------------------------------------------------------------
// Fused GEMM + tail. Block = 128 threads (4 waves), 64 samples per block.
// --------------------------------------------------------------------------
__global__ __launch_bounds__(128)
void layerstacks_fused_kernel(const float* __restrict__ x,
                              const unsigned short* __restrict__ whi_u,
                              const unsigned short* __restrict__ wlo_u,
                              const int*   __restrict__ ls_indices,
                              const float* __restrict__ b1,
                              const float* __restrict__ bf_,
                              const float* __restrict__ W2,
                              const float* __restrict__ b2,
                              const float* __restrict__ Wo,
                              const float* __restrict__ bo,
                              float* __restrict__ out) {
  __shared__ float l1buf[ROWS_PER_BLOCK * LDS_STRIDE];
  __shared__ float partial[128];

  const __bf16* whi = reinterpret_cast<const __bf16*>(whi_u);
  const __bf16* wlo = reinterpret_cast<const __bf16*>(wlo_u);

  const int tid    = threadIdx.x;
  const int wave   = tid >> 5;
  const int lane   = tid & 31;
  const int laneHi = lane >> 4;     // which half of the wave
  const int col16  = lane & 15;

  const int mBlock = blockIdx.x * ROWS_PER_BLOCK;
  const int rowA   = mBlock + wave * 16 + col16;   // A row this lane feeds
  const float* ap  = x + (size_t)rowA * K_DIM;

  v8f acc[N_TILES] = {};            // f32 accumulators, all 9 N-tiles

  for (int k0 = 0; k0 < K_DIM; k0 += 32) {
    // prefetch next A chunk (global_prefetch_b8)
    __builtin_prefetch(ap + k0 + 32, 0, 0);

    // ---- A fragment 16x32 bf16, split hi/lo (contiguous K per lane half, same order as B, wave32) ----
    // lane<16: K = k0+[0..15]; lane>=16: K = k0+[16..31]
    const int run1 = k0 + laneHi * 16;
    const int run2 = run1 + 8;
    v4f a0 = *(const v4f*)(ap + run1);
    v4f a1 = *(const v4f*)(ap + run1 + 4);
    v4f a2 = *(const v4f*)(ap + run2);
    v4f a3 = *(const v4f*)(ap + run2 + 4);

    float afl[16];
    #pragma unroll
    for (int i = 0; i < 4; ++i) {
      afl[i] = a0[i]; afl[4 + i] = a1[i]; afl[8 + i] = a2[i]; afl[12 + i] = a3[i];
    }

    // hi = truncate-to-bf16 (exact residual captured by lo); pack 2 elems per
    // VGPR with a single v_perm_b32 each.
    v8u hip, lop;
    #pragma unroll
    for (int j = 0; j < 8; ++j) {
      const unsigned f0 = __builtin_bit_cast(unsigned, afl[2 * j]);
      const unsigned f1 = __builtin_bit_cast(unsigned, afl[2 * j + 1]);
      hip[j] = __builtin_amdgcn_perm(f1, f0, BF16_PACK_SEL);
      const float h0 = __builtin_bit_cast(float, f0 & 0xFFFF0000u);
      const float h1 = __builtin_bit_cast(float, f1 & 0xFFFF0000u);
      const unsigned l0 = __builtin_bit_cast(unsigned, afl[2 * j]     - h0);
      const unsigned l1 = __builtin_bit_cast(unsigned, afl[2 * j + 1] - h1);
      lop[j] = __builtin_amdgcn_perm(l1, l0, BF16_PACK_SEL);
    }
    const v16bf ahi = __builtin_bit_cast(v16bf, hip);
    const v16bf alo = __builtin_bit_cast(v16bf, lop);

    // ---- B fragments 32x16 bf16 per N-tile; 3-product split accumulate ----
    // lane<16: K = k0+[0..15]; lane>=16: K = k0+[16..31]; column = lane&15
    const int kb = k0 + laneHi * 16;
    #pragma unroll
    for (int t = 0; t < N_TILES; ++t) {
      const size_t off = (size_t)(t * 16 + col16) * K_DIM + kb;
      v16bf bhi = *(const v16bf*)(whi + off);
      v16bf blo = *(const v16bf*)(wlo + off);
      acc[t] = __builtin_amdgcn_wmma_f32_16x16x32_bf16(
          false, ahi, false, bhi, (short)0, acc[t], false, false);
      acc[t] = __builtin_amdgcn_wmma_f32_16x16x32_bf16(
          false, ahi, false, blo, (short)0, acc[t], false, false);
      acc[t] = __builtin_amdgcn_wmma_f32_16x16x32_bf16(
          false, alo, false, bhi, (short)0, acc[t], false, false);
    }
  }

  // ---- Epilogue: spill C tiles to LDS (C layout: VGPR r -> M=r / M=8+r) ----
  {
    const int mOff = wave * 16 + laneHi * 8;
    #pragma unroll
    for (int t = 0; t < N_TILES; ++t) {
      #pragma unroll
      for (int r = 0; r < 8; ++r)
        l1buf[(mOff + r) * LDS_STRIDE + t * 16 + col16] = acc[t][r];
    }
  }
  __syncthreads();

  // ---- Stage 2: tail, 2 threads per sample (each does 32 of 64 L2 outs) ----
  const int s    = tid & 63;
  const int half = tid >> 6;
  const int gs   = mBlock + s;
  const int idx  = ls_indices[gs] & 7;
  const float* lrow = &l1buf[s * LDS_STRIDE];

  float l1x[30];
  #pragma unroll
  for (int j = 0; j < 15; ++j) {
    float v = (lrow[idx * 16 + j] + b1[idx * 16 + j]) +
              (lrow[128 + j]      + bf_[j]);
    l1x[j]      = clamp01(v * v * (127.0f / 128.0f));
    l1x[15 + j] = clamp01(v);
  }
  const float l1c15 = lrow[idx * 16 + 15] + b1[idx * 16 + 15];
  const float l1f15 = lrow[128 + 15]      + bf_[15];

  const float* w2  = W2 + (size_t)idx * 64 * 30;
  const float* wo  = Wo + idx * 64;
  const float* bb2 = b2 + idx * 64;

  float part = 0.0f;
  const int o0 = half * 32;
  for (int o = o0; o < o0 + 32; ++o) {
    float a = bb2[o];
    #pragma unroll
    for (int i = 0; i < 30; ++i) a = fmaf(l1x[i], w2[o * 30 + i], a);
    part += clamp01(a) * wo[o];
  }
  partial[tid] = part;
  __syncthreads();

  if (half == 0)
    out[gs] = partial[s] + partial[s + 64] + bo[idx] + l1f15 + l1c15;
}

// --------------------------------------------------------------------------
extern "C" void kernel_launch(void* const* d_in, const int* in_sizes, int n_in,
                              void* d_out, int out_size, void* d_ws, size_t ws_size,
                              hipStream_t stream) {
  (void)in_sizes; (void)n_in; (void)out_size; (void)ws_size;

  const float* x   = (const float*)d_in[0];
  const int*   lsi = (const int*)  d_in[1];
  const float* W1  = (const float*)d_in[2];
  const float* b1  = (const float*)d_in[3];
  const float* Wf  = (const float*)d_in[4];
  const float* bf_ = (const float*)d_in[5];
  const float* W2  = (const float*)d_in[6];
  const float* b2  = (const float*)d_in[7];
  const float* Wo  = (const float*)d_in[8];
  const float* bo  = (const float*)d_in[9];
  float* out = (float*)d_out;

  // d_ws: hi plane then lo plane of packed weights (bf16, 144x1536 each)
  unsigned short* whi = (unsigned short*)d_ws;
  unsigned short* wlo = whi + (size_t)N_COLS * K_DIM;   // +442368 bytes total

  {
    int total = N_COLS * K_DIM;
    dim3 blk(256), grd((total + 255) / 256);
    pack_weights_kernel<<<grd, blk, 0, stream>>>(W1, Wf, whi, wlo);
  }
  {
    dim3 blk(128), grd(M_TOTAL / ROWS_PER_BLOCK);
    layerstacks_fused_kernel<<<grd, blk, 0, stream>>>(
        x, whi, wlo, lsi, b1, bf_, W2, b2, Wo, bo, out);
  }
}
